// UpperBound_kmeans_12034498363598
// MI455X (gfx1250) — matrix operand order repaired and checked by reading, not compiled, not run on hardware
//
#include <hip/hip_runtime.h>
#include <hip/hip_bf16.h>

#define K_CLASSES 100000
#define FEAT 128
#define BATCH 1024

#define CSQ_GRID 128
#define GEMM_GRID_X 128   // x: column-tile stride ; y: 8 row-blocks of 128 rows
#define ROWBLK 128

typedef __attribute__((ext_vector_type(16))) __bf16 v16bf;
typedef __attribute__((ext_vector_type(8)))  float  v8f;

union BF16Frag { v16bf v; unsigned short s[16]; uint4 q[2]; };

// Truncation split: f ~= hi + lo with ~16 mantissa bits total; dropping lo*lo in
// the GEMM leaves ~2^-16 relative error -> argmin-safe.
__device__ __forceinline__ void split_bf16(float f, unsigned short& hi, unsigned short& lo) {
    unsigned u = __float_as_uint(f);
    hi = (unsigned short)(u >> 16);
    float hf = __uint_as_float(u & 0xFFFF0000u);
    lo = (unsigned short)(__float_as_uint(f - hf) >> 16);
}

// ---------------------------------------------------------------------------
// Kernel 1: init packed argmin slots
// ---------------------------------------------------------------------------
__global__ void k_init(unsigned long long* __restrict__ best) {
    int t = blockIdx.x * blockDim.x + threadIdx.x;
    if (t < BATCH) best[t] = ~0ull;
}

// ---------------------------------------------------------------------------
// Kernel 2: c_sq[k], partial column sums; optionally pre-split centers into
// bf16 hi/lo planes so the GEMM hot loop has zero conversion VALU.
// One wave per row: lane l handles feats 4l..4l+3 (coalesced 512B/row).
// ---------------------------------------------------------------------------
template<bool SPLIT>
__global__ __launch_bounds__(256) void k_csq(const float* __restrict__ centers,
                                             float* __restrict__ c_sq,
                                             float* __restrict__ csum_part,
                                             float* __restrict__ scsq_part,
                                             unsigned short* __restrict__ chi,
                                             unsigned short* __restrict__ clo) {
    __shared__ float4 red[256];
    __shared__ float  wsum[8];
    const int tid = threadIdx.x, wave = tid >> 5, lane = tid & 31;

    float4 cp = {0.f, 0.f, 0.f, 0.f};
    float  sc = 0.f;
    for (int k = blockIdx.x * 8 + wave; k < K_CLASSES; k += CSQ_GRID * 8) {
        float4 v = *reinterpret_cast<const float4*>(centers + (size_t)k * FEAT + lane * 4);
        if (SPLIT) {
            ushort4 h4, l4;
            split_bf16(v.x, h4.x, l4.x);
            split_bf16(v.y, h4.y, l4.y);
            split_bf16(v.z, h4.z, l4.z);
            split_bf16(v.w, h4.w, l4.w);
            *reinterpret_cast<ushort4*>(chi + (size_t)k * FEAT + lane * 4) = h4;
            *reinterpret_cast<ushort4*>(clo + (size_t)k * FEAT + lane * 4) = l4;
        }
        cp.x += v.x; cp.y += v.y; cp.z += v.z; cp.w += v.w;
        float ss = v.x * v.x + v.y * v.y + v.z * v.z + v.w * v.w;
        #pragma unroll
        for (int m = 16; m >= 1; m >>= 1) ss += __shfl_xor(ss, m, 32);
        if (lane == 0) { c_sq[k] = ss; sc += ss; }
    }
    red[tid] = cp;
    if (lane == 0) wsum[wave] = sc;
    __syncthreads();
    if (tid < 32) {
        float4 t = red[tid];
        #pragma unroll
        for (int w = 1; w < 8; ++w) {
            float4 o = red[tid + 32 * w];
            t.x += o.x; t.y += o.y; t.z += o.z; t.w += o.w;
        }
        float* dst = csum_part + (size_t)blockIdx.x * FEAT + tid * 4;
        dst[0] = t.x; dst[1] = t.y; dst[2] = t.z; dst[3] = t.w;
    }
    if (tid == 0) {
        float s = 0.f;
        #pragma unroll
        for (int w = 0; w < 8; ++w) s += wsum[w];
        scsq_part[blockIdx.x] = s;
    }
}

// ---------------------------------------------------------------------------
// Kernel 3: deterministic reduction of partials
// ---------------------------------------------------------------------------
__global__ void k_reduce(const float* __restrict__ csum_part,
                         const float* __restrict__ scsq_part,
                         float* __restrict__ csum,
                         float* __restrict__ sum_csq) {
    int d = threadIdx.x;
    if (d < FEAT) {
        float s = 0.f;
        for (int i = 0; i < CSQ_GRID; ++i) s += csum_part[(size_t)i * FEAT + d];
        csum[d] = s;
    }
    if (d == 0) {
        float s = 0.f;
        for (int i = 0; i < CSQ_GRID; ++i) s += scsq_part[i];
        *sum_csq = s;
    }
}

// ---------------------------------------------------------------------------
// Kernel 4: WMMA GEMM + fused per-row argmin of (||c_k||^2 - 2 x.c_k).
// Block = 256 thr (8 wave32). blockIdx.y selects 128 rows; wave w owns rows
// [.. + w*16, +16). ALL waves sweep the SAME 16-column tile each iteration,
// so their B loads hit identical addresses -> WGP$ dedup, L2 sees centers
// only 8x (0.41 GB) instead of 64x. A is register-cached split-bf16
// (contiguous K = 16h+s order, not the ISA table order); 3 bf16 WMMAs (hi*hi + hi*lo + lo*hi) per K=32 chunk.
// ---------------------------------------------------------------------------
template<bool PRESPLIT>
__global__ __launch_bounds__(256) void k_gemm_argmin(const float* __restrict__ x,
                                                     const float* __restrict__ centers,
                                                     const unsigned short* __restrict__ chi,
                                                     const unsigned short* __restrict__ clo,
                                                     const float* __restrict__ c_sq,
                                                     unsigned long long* __restrict__ best) {
    const int tid  = threadIdx.x;
    const int wave = tid >> 5;
    const int lane = tid & 31;
    const int h    = lane >> 4;   // lane half (K-subset selector)
    const int ln   = lane & 15;
    const int rowbase = blockIdx.y * ROWBLK + wave * 16;

    // ---- A fragments, loaded once. Lane L: row M=L%16; half h: v0..7 = K 16h+0..15
    // (2 packed bf16/VGPR, low half = even K).
    BF16Frag ah[4], al[4];
    {
        const float* xp = x + (size_t)(rowbase + ln) * FEAT;
        #pragma unroll
        for (int c = 0; c < 4; ++c) {
            const int k1 = c * 32 + 16 * h;
            float tmp[16];
            *reinterpret_cast<float4*>(&tmp[0])  = *reinterpret_cast<const float4*>(xp + k1);
            *reinterpret_cast<float4*>(&tmp[4])  = *reinterpret_cast<const float4*>(xp + k1 + 4);
            *reinterpret_cast<float4*>(&tmp[8])  = *reinterpret_cast<const float4*>(xp + k1 + 8);
            *reinterpret_cast<float4*>(&tmp[12]) = *reinterpret_cast<const float4*>(xp + k1 + 12);
            #pragma unroll
            for (int t = 0; t < 16; ++t) split_bf16(tmp[t], ah[c].s[t], al[c].s[t]);
        }
    }

    float    bval[8];
    unsigned bidx[8];
    #pragma unroll
    for (int j = 0; j < 8; ++j) { bval[j] = __builtin_inff(); bidx[j] = 0u; }

    const int NT = (K_CLASSES + 15) >> 4;   // 6250 column tiles of 16
    for (int t = blockIdx.x; t < NT; t += gridDim.x) {
        const int col  = t * 16 + ln;
        const int colc = (col < K_CLASSES) ? col : (K_CLASSES - 1);
        const float csq = (col < K_CLASSES) ? c_sq[col] : __builtin_inff();

        // Prefetch next column tile (emits global_prefetch_b8).
        const int coln = (t + (int)gridDim.x) * 16 + ln;
        if (coln < K_CLASSES) {
            if (PRESPLIT) __builtin_prefetch(chi + (size_t)coln * FEAT + h * 16, 0, 1);
            else          __builtin_prefetch(centers + (size_t)coln * FEAT + h * 16, 0, 1);
        }

        v8f acc = {0.f, 0.f, 0.f, 0.f, 0.f, 0.f, 0.f, 0.f};
        #pragma unroll
        for (int c = 0; c < 4; ++c) {
            // B frag (32x16 bf16), same contiguous K order as A: lane L: col N=L%16; half h: v_i = K 16h+2i,2i+1
            // -> lane needs feats [c*32 + 16h, +16) of column colc, contiguous.
            BF16Frag bh, bl;
            if (PRESPLIT) {
                const unsigned short* hp = chi + (size_t)colc * FEAT + c * 32 + h * 16;
                const unsigned short* lp = clo + (size_t)colc * FEAT + c * 32 + h * 16;
                bh.q[0] = *reinterpret_cast<const uint4*>(hp);
                bh.q[1] = *reinterpret_cast<const uint4*>(hp + 8);
                bl.q[0] = *reinterpret_cast<const uint4*>(lp);
                bl.q[1] = *reinterpret_cast<const uint4*>(lp + 8);
            } else {
                const float* bp = centers + (size_t)colc * FEAT + c * 32 + h * 16;
                float tmp[16];
                *reinterpret_cast<float4*>(&tmp[0])  = *reinterpret_cast<const float4*>(bp);
                *reinterpret_cast<float4*>(&tmp[4])  = *reinterpret_cast<const float4*>(bp + 4);
                *reinterpret_cast<float4*>(&tmp[8])  = *reinterpret_cast<const float4*>(bp + 8);
                *reinterpret_cast<float4*>(&tmp[12]) = *reinterpret_cast<const float4*>(bp + 12);
                #pragma unroll
                for (int tt = 0; tt < 16; ++tt) split_bf16(tmp[tt], bh.s[tt], bl.s[tt]);
            }
            // 3xBF16 ~ FP32: hi*hi + hi*lo + lo*hi, f32 accumulate
            acc = __builtin_amdgcn_wmma_f32_16x16x32_bf16(false, ah[c].v, false, bh.v, (short)0, acc, false, false);
            acc = __builtin_amdgcn_wmma_f32_16x16x32_bf16(false, ah[c].v, false, bl.v, (short)0, acc, false, false);
            acc = __builtin_amdgcn_wmma_f32_16x16x32_bf16(false, al[c].v, false, bh.v, (short)0, acc, false, false);
        }

        // C/D layout: VGPR j -> row M = j + 8h ; lane -> col N = lane%16.
        #pragma unroll
        for (int j = 0; j < 8; ++j) {
            float s = csq - 2.0f * acc[j];   // x_sq is per-row constant: argmin-invariant
            if (s < bval[j]) { bval[j] = s; bidx[j] = (unsigned)col; }
        }
    }

    // Cross-lane argmin within each 16-lane group (each row lives in one half).
    #pragma unroll
    for (int m = 1; m < 16; m <<= 1) {
        #pragma unroll
        for (int j = 0; j < 8; ++j) {
            float    ov = __shfl_xor(bval[j], m, 16);
            unsigned oi = (unsigned)__shfl_xor((int)bidx[j], m, 16);
            if (ov < bval[j] || (ov == bval[j] && oi < bidx[j])) { bval[j] = ov; bidx[j] = oi; }
        }
    }
    if (ln == 0) {
        #pragma unroll
        for (int j = 0; j < 8; ++j) {
            unsigned b32 = __float_as_uint(bval[j]);
            unsigned key = (b32 & 0x80000000u) ? ~b32 : (b32 | 0x80000000u); // order-preserving
            unsigned long long packed = ((unsigned long long)key << 32) | (unsigned long long)bidx[j];
            atomicMin(&best[rowbase + j + 8 * h], packed);
        }
    }
}

// ---------------------------------------------------------------------------
// Kernel 5: finalize. One block, 32 waves; wave w handles rows w, w+32, ...
// i_dist gather dot + analytic row_sum; deterministic tree reduction.
// ---------------------------------------------------------------------------
__global__ __launch_bounds__(1024) void k_finalize(const float* __restrict__ x,
                                                   const float* __restrict__ centers,
                                                   const int* __restrict__ labels,
                                                   const float* __restrict__ c_sq,
                                                   const float* __restrict__ csum,
                                                   const float* __restrict__ sum_csq,
                                                   const unsigned long long* __restrict__ best,
                                                   float* __restrict__ out) {
    __shared__ float lp[BATCH];
    __shared__ int   af[BATCH];
    const int tid = threadIdx.x, wave = tid >> 5, lane = tid & 31;
    const float scsq = *sum_csq;
    float4 cs = *reinterpret_cast<const float4*>(csum + lane * 4);

    for (int b = wave; b < BATCH; b += 32) {
        float4 xv = *reinterpret_cast<const float4*>(x + (size_t)b * FEAT + lane * 4);
        int lab = labels[b];
        if (lab < 0) lab = 0; if (lab >= K_CLASSES) lab = K_CLASSES - 1;
        float4 cv = *reinterpret_cast<const float4*>(centers + (size_t)lab * FEAT + lane * 4);
        float p0 = xv.x * xv.x + xv.y * xv.y + xv.z * xv.z + xv.w * xv.w;  // ||x||^2
        float p1 = xv.x * cs.x + xv.y * cs.y + xv.z * cs.z + xv.w * cs.w;  // x . csum
        float p2 = xv.x * cv.x + xv.y * cv.y + xv.z * cv.z + xv.w * cv.w;  // x . c_label
        #pragma unroll
        for (int m = 16; m >= 1; m >>= 1) {
            p0 += __shfl_xor(p0, m, 32);
            p1 += __shfl_xor(p1, m, 32);
            p2 += __shfl_xor(p2, m, 32);
        }
        if (lane == 0) {
            float i_dist  = p0 + c_sq[lab] - 2.0f * p2;
            float row_sum = (float)K_CLASSES * p0 + scsq - 2.0f * p1;
            lp[b] = i_dist - (row_sum - i_dist) * (1.0f / (3.0f * (float)(K_CLASSES - 1)));
            unsigned idx = (unsigned)(best[b] & 0xffffffffull);
            af[b] = (idx == (unsigned)lab) ? 1 : 0;
        }
    }
    __syncthreads();
    #pragma unroll
    for (int s = BATCH / 2; s > 0; s >>= 1) {
        if (tid < s) { lp[tid] += lp[tid + s]; af[tid] += af[tid + s]; }
        __syncthreads();
    }
    if (tid == 0) {
        out[0] = lp[0] * (1.0f / (float)BATCH);
        out[1] = (float)af[0];
    }
}

// ---------------------------------------------------------------------------
extern "C" void kernel_launch(void* const* d_in, const int* in_sizes, int n_in,
                              void* d_out, int out_size, void* d_ws, size_t ws_size,
                              hipStream_t stream) {
    const float* x       = (const float*)d_in[0];
    const float* centers = (const float*)d_in[1];
    const int*   labels  = (const int*)d_in[2];
    float*       out     = (float*)d_out;

    // Workspace layout
    const size_t OFF_BEST = 0;                                    // 8 KB
    const size_t OFF_CSQ  = 8192;                                 // 400 KB
    const size_t OFF_CPRT = OFF_CSQ  + (size_t)K_CLASSES * 4;     // 64 KB
    const size_t OFF_SPRT = OFF_CPRT + (size_t)CSQ_GRID * FEAT * 4;
    const size_t OFF_CSUM = OFF_SPRT + (size_t)CSQ_GRID * 4;
    const size_t OFF_SCSQ = OFF_CSUM + (size_t)FEAT * 4;
    const size_t OFF_CHI  = (OFF_SCSQ + 4 + 255) & ~(size_t)255;  // bf16 hi plane, 25.6 MB
    const size_t OFF_CLO  = OFF_CHI + (size_t)K_CLASSES * FEAT * 2;
    const size_t NEED     = OFF_CLO + (size_t)K_CLASSES * FEAT * 2;

    char* ws = (char*)d_ws;
    unsigned long long* best = (unsigned long long*)(ws + OFF_BEST);
    float* c_sq      = (float*)(ws + OFF_CSQ);
    float* csum_part = (float*)(ws + OFF_CPRT);
    float* scsq_part = (float*)(ws + OFF_SPRT);
    float* csum      = (float*)(ws + OFF_CSUM);
    float* sum_csq   = (float*)(ws + OFF_SCSQ);
    unsigned short* chi = (unsigned short*)(ws + OFF_CHI);
    unsigned short* clo = (unsigned short*)(ws + OFF_CLO);

    const bool presplit = (ws_size >= NEED);   // constant per harness -> deterministic

    k_init<<<4, 256, 0, stream>>>(best);
    if (presplit) {
        k_csq<true><<<CSQ_GRID, 256, 0, stream>>>(centers, c_sq, csum_part, scsq_part, chi, clo);
        k_reduce<<<1, 128, 0, stream>>>(csum_part, scsq_part, csum, sum_csq);
        k_gemm_argmin<true><<<dim3(GEMM_GRID_X, BATCH / ROWBLK), 256, 0, stream>>>(x, centers, chi, clo, c_sq, best);
    } else {
        k_csq<false><<<CSQ_GRID, 256, 0, stream>>>(centers, c_sq, csum_part, scsq_part, chi, clo);
        k_reduce<<<1, 128, 0, stream>>>(csum_part, scsq_part, csum, sum_csq);
        k_gemm_argmin<false><<<dim3(GEMM_GRID_X, BATCH / ROWBLK), 256, 0, stream>>>(x, centers, chi, clo, c_sq, best);
    }
    k_finalize<<<1, 1024, 0, stream>>>(x, centers, labels, c_sq, csum, sum_csq, best, out);
}
